// PointNetSetAbstraction_712964571702
// MI455X (gfx1250) — compile-verified, matrix operand order checked by reading, not run on hardware
//
#include <hip/hip_runtime.h>
#include <hip/hip_bf16.h>
#include <math.h>

// ---------------------------------------------------------------------------
// PointNet++ Set Abstraction for MI455X (gfx1250, wave32, WMMA)
//   B=16, N=4096, NPOINT=1024, NSAMPLE=32, RADIUS=0.2, IN_CH=64, MLP=[64,64,128]
// Pipeline:
//   K1 fps_kernel            : 1 block/batch, sequential FPS, emits new_xyz
//   K2 transpose_points      : points [B,64,N] f32 -> ptsT [B,N,64] f16
//   K3 prep_weights          : W f32 [Cin,Cout] -> Wt f16 [Cout,CinPad] (padded)
//   K4 ballquery_gather      : wave/center, ballot-ordered first-32-in-radius,
//                              writes A0 f16 [B*S*K, 96] (feats | relxyz | 0pad)
//   K5 gemm<96,64,noBN>      : WMMA f16->f32, Y0 f16 + per-block channel stats
//   K6 finalize_stats        : stats -> (scale,shift) for BN0
//   K7 gemm<64,64,BN0>       : BN+ReLU applied on A-load, Y1 f16 + stats
//   K8 finalize_stats        : BN1
//   K9 gemm<64,128,BN1>      : Y2 f16 + stats
//   K10 finalize_stats       : BN2
//   K11 maxpool              : relu(BN2(Y2)) max over K=32, LDS-transposed store
// Bias b_l is mathematically cancelled by BN mean-subtraction -> skipped.
// All reductions are fixed-order (no float atomics) -> deterministic replays.
// ---------------------------------------------------------------------------

typedef __attribute__((ext_vector_type(16))) _Float16 v16h;
typedef __attribute__((ext_vector_type(8)))  _Float16 v8h;
typedef __attribute__((ext_vector_type(8)))  float    v8f;

#define BQ    16
#define NQ    4096
#define SQ    1024
#define KQ    32
#define R2Q   0.04f
#define ROWS  (BQ * SQ * KQ)       // 524288
#define MT    (ROWS / 16)          // 32768 M-tiles
#define C0P   96                   // 67 padded to 96
#define GEMM_BLOCKS 1024

// ---------------- workspace layout (bytes, all 256B-aligned) ----------------
#define OFF_NEWXYZ  ((size_t)0)                       // 16*1024*3 f32  = 196608
#define OFF_PTST    ((size_t)196608)                  // 16*4096*64 f16 = 8388608
#define OFF_W0T     ((size_t)8585216)                 // 64*96  f16     = 12288
#define OFF_W1T     ((size_t)8597504)                 // 64*64  f16     = 8192
#define OFF_W2T     ((size_t)8605696)                 // 128*64 f16     = 16384
#define OFF_SC      ((size_t)8622080)                 // 6*128 f32      = 3072
#define OFF_P0      ((size_t)8625152)                 // 1024*2*64 f32  = 524288
#define OFF_P1      ((size_t)9149440)                 // 1024*2*64 f32  = 524288
#define OFF_P2      ((size_t)9673728)                 // 1024*2*128 f32 = 1048576
#define OFF_A0      ((size_t)10722304)                // 524288*96 f16  = 100663296 (Y1 aliases)
#define OFF_Y0      ((size_t)111385600)               // 524288*64 f16  = 67108864
#define OFF_Y2      ((size_t)178494464)               // 524288*128 f16 = 134217728
#define WS_NEEDED   ((size_t)312712192)

// ---------------------------------------------------------------------------
// K1: farthest point sampling. One block per batch. dist held in registers
// (4 per thread), xyz cached in LDS, two-level wave32 shuffle argmax with
// first-occurrence tie-break (matches jnp.argmax).
// ---------------------------------------------------------------------------
__global__ __launch_bounds__(1024) void fps_kernel(const float* __restrict__ xyz,
                                                   float* __restrict__ newxyz_ws,
                                                   float* __restrict__ out_newxyz) {
  __shared__ float sx[NQ], sy[NQ], sz[NQ];
  __shared__ float s_rv[32];
  __shared__ int   s_ri[32];
  __shared__ int   s_far;
  const int b = blockIdx.x;
  const float* xb = xyz + (size_t)b * 3 * NQ;
  for (int i = threadIdx.x; i < NQ; i += 1024) {
    sx[i] = xb[i];
    sy[i] = xb[NQ + i];
    sz[i] = xb[2 * NQ + i];
  }
  if (threadIdx.x == 0) s_far = 0;
  float dreg[4] = {1e10f, 1e10f, 1e10f, 1e10f};
  __syncthreads();

  const int lane = threadIdx.x & 31;
  const int wid  = threadIdx.x >> 5;
  for (int t = 0; t < SQ; ++t) {
    const int far = s_far;
    const float cx = sx[far], cy = sy[far], cz = sz[far];
    if (threadIdx.x == 0) {
      size_t sg = (size_t)b * SQ + t;
      newxyz_ws[sg * 3 + 0] = cx;
      newxyz_ws[sg * 3 + 1] = cy;
      newxyz_ws[sg * 3 + 2] = cz;
      out_newxyz[((size_t)b * 3 + 0) * SQ + t] = cx;
      out_newxyz[((size_t)b * 3 + 1) * SQ + t] = cy;
      out_newxyz[((size_t)b * 3 + 2) * SQ + t] = cz;
    }
    float bv = -1.0f; int bi = 0x7fffffff;
#pragma unroll
    for (int j = 0; j < 4; ++j) {
      const int i = threadIdx.x + j * 1024;
      const float dx = sx[i] - cx, dy = sy[i] - cy, dz = sz[i] - cz;
      const float d  = dx * dx + dy * dy + dz * dz;
      const float nd = fminf(dreg[j], d);
      dreg[j] = nd;
      if (nd > bv || (nd == bv && i < bi)) { bv = nd; bi = i; }
    }
#pragma unroll
    for (int off = 16; off > 0; off >>= 1) {
      const float ov = __shfl_xor(bv, off);
      const int   oi = __shfl_xor(bi, off);
      if (ov > bv || (ov == bv && oi < bi)) { bv = ov; bi = oi; }
    }
    if (lane == 0) { s_rv[wid] = bv; s_ri[wid] = bi; }
    __syncthreads();
    if (wid == 0) {
      bv = s_rv[lane]; bi = s_ri[lane];
#pragma unroll
      for (int off = 16; off > 0; off >>= 1) {
        const float ov = __shfl_xor(bv, off);
        const int   oi = __shfl_xor(bi, off);
        if (ov > bv || (ov == bv && oi < bi)) { bv = ov; bi = oi; }
      }
      if (lane == 0) s_far = bi;
    }
    __syncthreads();
  }
}

// ---------------------------------------------------------------------------
// K2: points [B,64,N] f32 -> ptsT [B,N,64] f16 (coalesced reads)
// ---------------------------------------------------------------------------
__global__ __launch_bounds__(256) void transpose_points_kernel(const float* __restrict__ pts,
                                                               _Float16* __restrict__ ptsT) {
  const size_t t = (size_t)blockIdx.x * 256 + threadIdx.x;
  if (t >= (size_t)BQ * 64 * NQ) return;
  const int n = (int)(t & (NQ - 1));
  const size_t bc = t >> 12;
  const int c = (int)(bc & 63);
  const int b = (int)(bc >> 6);
  ptsT[((size_t)b * NQ + n) * 64 + c] = (_Float16)pts[t];
}

// ---------------------------------------------------------------------------
// K3: transpose+pad weights to f16 Wt[Cout][CinPad].
// A0 column map: c<64 -> input channel 3+c (feature), c in 64..66 -> rel xyz,
// c>=67 -> zero. W0 is [67,64], W1 [64,64], W2 [64,128].
// ---------------------------------------------------------------------------
__global__ __launch_bounds__(256) void prep_weights_kernel(const float* __restrict__ W0,
                                                           const float* __restrict__ W1,
                                                           const float* __restrict__ W2,
                                                           _Float16* __restrict__ W0t,
                                                           _Float16* __restrict__ W1t,
                                                           _Float16* __restrict__ W2t) {
  const int t = threadIdx.x + blockIdx.x * 256;
  const int stride = 256 * gridDim.x;
  for (int i = t; i < 64 * C0P; i += stride) {
    const int d = i / C0P, c = i % C0P;
    float v = 0.0f;
    if (c < 64)      v = W0[(3 + c) * 64 + d];
    else if (c < 67) v = W0[(c - 64) * 64 + d];
    W0t[i] = (_Float16)v;
  }
  for (int i = t; i < 64 * 64; i += stride) {
    const int d = i / 64, c = i % 64;
    W1t[i] = (_Float16)W1[c * 64 + d];
  }
  for (int i = t; i < 128 * 64; i += stride) {
    const int d = i / 64, c = i % 64;
    W2t[i] = (_Float16)W2[c * 128 + d];
  }
}

// ---------------------------------------------------------------------------
// K4: ball query (first 32 in-radius by index, pad with first) + gather into
// A0 rows: [f0..f63 | relx rely relz 0.. | 0 pad to 96] as f16. Wave/center.
// ---------------------------------------------------------------------------
__global__ __launch_bounds__(256) void ballquery_gather_kernel(const float* __restrict__ xyz,
                                                               const _Float16* __restrict__ ptsT,
                                                               const float* __restrict__ newxyz_ws,
                                                               _Float16* __restrict__ A0) {
  __shared__ int sidx[8][32];
  const int wid  = threadIdx.x >> 5;
  const int lane = threadIdx.x & 31;
  const int sg   = blockIdx.x * 8 + wid;         // 0 .. B*S-1
  const int b    = sg >> 10;
  const float* xb = xyz + (size_t)b * 3 * NQ;
  const float cx = newxyz_ws[(size_t)sg * 3 + 0];
  const float cy = newxyz_ws[(size_t)sg * 3 + 1];
  const float cz = newxyz_ws[(size_t)sg * 3 + 2];

  int found = 0;
  for (int base = 0; base < NQ && found < KQ; base += 32) {
    const int n = base + lane;
    const float dx = xb[n] - cx, dy = xb[NQ + n] - cy, dz = xb[2 * NQ + n] - cz;
    const float d2 = dx * dx + dy * dy + dz * dz;
    const bool in = (d2 <= R2Q);                  // reference keeps sqr <= r^2
    const unsigned mask = (unsigned)__ballot(in);
    if (in) {
      const int slot = found + __popc(mask & ((1u << lane) - 1u));
      if (slot < KQ) sidx[wid][slot] = n;
    }
    found += __popc(mask);
  }
  const int fc = found < KQ ? found : KQ;         // fc >= 1 (center is in set)
  const int first = sidx[wid][0];
  if (lane >= fc) sidx[wid][lane] = first;
  const int pidx = sidx[wid][lane];

  _Float16* row = A0 + ((size_t)sg * KQ + lane) * C0P;
  const v8h* src = (const v8h*)(ptsT + ((size_t)b * NQ + pidx) * 64);
  v8h* dst = (v8h*)row;
#pragma unroll
  for (int j = 0; j < 8; ++j) dst[j] = src[j];    // features, 8 x 16B
  const float rx = xb[pidx] - cx, ry = xb[NQ + pidx] - cy, rz = xb[2 * NQ + pidx] - cz;
  v8h tail = {};
  tail[0] = (_Float16)rx; tail[1] = (_Float16)ry; tail[2] = (_Float16)rz;
  dst[8] = tail;
  const v8h zv = {};
  dst[9] = zv; dst[10] = zv; dst[11] = zv;        // pad cols 72..95
}

// ---------------------------------------------------------------------------
// Fragment loader per CDNA5 16-bit A/B 16x32 layout:
// lane L: row M = L%16, K halves at kb..kb+7 and kb+16..kb+23, kb = ktile + 8*(L>=16).
// Optionally applies BN (y*scale+shift) + ReLU on load (for A of layers 1,2).
// ---------------------------------------------------------------------------
template <bool BN>
__device__ inline v16h load_frag(const _Float16* __restrict__ mat, int row_base, int ld,
                                 int ktile, int lane,
                                 const float* __restrict__ sc, const float* __restrict__ sh) {
  const int m  = lane & 15;
  const int kb = ktile + ((lane >> 4) << 3);
  const _Float16* p = mat + (size_t)(row_base + m) * ld + kb;
  const v8h lo = *reinterpret_cast<const v8h*>(p);
  const v8h hi = *reinterpret_cast<const v8h*>(p + 16);
  v16h r;
  if (BN) {
#pragma unroll
    for (int j = 0; j < 8; ++j) {
      const int c0 = kb + j, c1 = kb + 16 + j;
      float a = fmaxf(fmaf((float)lo[j], sc[c0], sh[c0]), 0.0f);
      float b = fmaxf(fmaf((float)hi[j], sc[c1], sh[c1]), 0.0f);
      r[j] = (_Float16)a;
      r[8 + j] = (_Float16)b;
    }
  } else {
#pragma unroll
    for (int j = 0; j < 8; ++j) { r[j] = lo[j]; r[8 + j] = hi[j]; }
  }
  return r;
}

// ---------------------------------------------------------------------------
// K5/K7/K9: GEMM Y = act(A) x Wt^T via v_wmma_f32_16x16x32_f16.
// 8 waves/block, grid-stride over M-tiles, all COUT tiles per wave.
// Emits deterministic per-block channel sum/sumsq partials for BN.
// ---------------------------------------------------------------------------
template <int CIN, int COUT, bool BN>
__global__ __launch_bounds__(256) void gemm_bn_kernel(const _Float16* __restrict__ A,
                                                      const _Float16* __restrict__ Wt,
                                                      _Float16* __restrict__ Y,
                                                      const float* __restrict__ scale,
                                                      const float* __restrict__ shift,
                                                      float* __restrict__ partial,  // [blocks][2][COUT]
                                                      int mtiles) {
  constexpr int NT = COUT / 16;
  constexpr int KT = CIN / 32;
  __shared__ float s_sc[128];
  __shared__ float s_sh[128];
  __shared__ float s_part[8][2][COUT];
  const int wid  = threadIdx.x >> 5;
  const int lane = threadIdx.x & 31;
  if (BN) {
    for (int c = threadIdx.x; c < CIN; c += 256) { s_sc[c] = scale[c]; s_sh[c] = shift[c]; }
    __syncthreads();
  }

  float stat_s[NT], stat_q[NT];
#pragma unroll
  for (int n = 0; n < NT; ++n) { stat_s[n] = 0.0f; stat_q[n] = 0.0f; }

  const int wstep = gridDim.x * 8;
  for (int tile = blockIdx.x * 8 + wid; tile < mtiles; tile += wstep) {
    const int row0 = tile * 16;
    // prefetch next tile's A slice (gfx1250 global_prefetch_b8)
    if (tile + wstep < mtiles)
      __builtin_prefetch(A + (size_t)(tile + wstep) * 16 * CIN + lane * ((16 * CIN) / 32), 0, 1);

    v8f acc[NT];
#pragma unroll
    for (int n = 0; n < NT; ++n)
#pragma unroll
      for (int r = 0; r < 8; ++r) acc[n][r] = 0.0f;

#pragma unroll
    for (int k = 0; k < KT; ++k) {
      const v16h a = load_frag<BN>(A, row0, CIN, k * 32, lane, s_sc, s_sh);
#pragma unroll
      for (int n = 0; n < NT; ++n) {
        const v16h b = load_frag<false>(Wt, n * 16, CIN, k * 32, lane, nullptr, nullptr);
        acc[n] = __builtin_amdgcn_wmma_f32_16x16x32_f16(false, a, false, b,
                                                        (short)0, acc[n], false, false);
      }
    }
    // store f16 + per-lane channel stats (lane holds column N = lane%16; rows M = r + 8*(lane>=16))
    const int mOff  = (lane >> 4) << 3;
    const int nLane = lane & 15;
#pragma unroll
    for (int n = 0; n < NT; ++n) {
      float s = 0.0f, q = 0.0f;
#pragma unroll
      for (int r = 0; r < 8; ++r) {
        const float v = acc[n][r];
        s += v; q += v * v;
        Y[(size_t)(row0 + mOff + r) * COUT + n * 16 + nLane] = (_Float16)v;
      }
      stat_s[n] += s; stat_q[n] += q;
    }
  }

  // lanes L and L+16 share channel -> combine, lanes 0..15 publish per-wave partials
#pragma unroll
  for (int n = 0; n < NT; ++n) {
    const float s = stat_s[n] + __shfl_xor(stat_s[n], 16);
    const float q = stat_q[n] + __shfl_xor(stat_q[n], 16);
    if (lane < 16) { s_part[wid][0][n * 16 + lane] = s; s_part[wid][1][n * 16 + lane] = q; }
  }
  __syncthreads();
  for (int c = threadIdx.x; c < COUT; c += 256) {
    float s = 0.0f, q = 0.0f;
#pragma unroll
    for (int w = 0; w < 8; ++w) { s += s_part[w][0][c]; q += s_part[w][1][c]; }
    partial[(size_t)blockIdx.x * 2 * COUT + c] = s;
    partial[(size_t)blockIdx.x * 2 * COUT + COUT + c] = q;
  }
}

// ---------------------------------------------------------------------------
// K6/K8/K10: fold partials -> BN scale/shift (fixed-order, deterministic).
// scale = g * rsqrt(var+eps), shift = be - mean*scale
// ---------------------------------------------------------------------------
__global__ void finalize_stats_kernel(const float* __restrict__ partial, int nblocks, int cout,
                                      const float* __restrict__ g, const float* __restrict__ be,
                                      float* __restrict__ scale, float* __restrict__ shift) {
  const int c = threadIdx.x;
  if (c >= cout) return;
  float s = 0.0f, q = 0.0f;
  for (int b = 0; b < nblocks; ++b) {
    s += partial[(size_t)b * 2 * cout + c];
    q += partial[(size_t)b * 2 * cout + cout + c];
  }
  const float inv  = 1.0f / (float)ROWS;
  const float mean = s * inv;
  const float var  = q * inv - mean * mean;
  const float sc   = g[c] * rsqrtf(var + 1e-5f);
  scale[c] = sc;
  shift[c] = be[c] - mean * sc;
}

// ---------------------------------------------------------------------------
// K11: out[b,d,s] = max_k relu(BN2(Y2[b,s,k,d])). LDS transpose for coalesced
// output stores. One block per (b, 32-center chunk).
// ---------------------------------------------------------------------------
__global__ __launch_bounds__(256) void maxpool_kernel(const _Float16* __restrict__ Y2,
                                                      const float* __restrict__ scale,
                                                      const float* __restrict__ shift,
                                                      float* __restrict__ out_points) {
  __shared__ float tile[32][129];
  const int b  = blockIdx.x >> 5;
  const int s0 = (blockIdx.x & 31) * 32;
  for (int id = threadIdx.x; id < 32 * 128; id += 256) {
    const int d  = id & 127;
    const int sl = id >> 7;
    const size_t row = ((size_t)(b * SQ + s0 + sl)) * KQ;
    const float sc = scale[d], sh = shift[d];
    float m = 0.0f;                      // relu output >= 0
    for (int k = 0; k < KQ; ++k) {
      float v = (float)Y2[(row + k) * 128 + d];
      v = fmaxf(fmaf(v, sc, sh), 0.0f);
      m = fmaxf(m, v);
    }
    tile[sl][d] = m;
  }
  __syncthreads();
  for (int id = threadIdx.x; id < 32 * 128; id += 256) {
    const int sl = id & 31;
    const int d  = id >> 5;
    out_points[((size_t)b * 128 + d) * SQ + s0 + sl] = tile[sl][d];
  }
}

// ---------------------------------------------------------------------------
extern "C" void kernel_launch(void* const* d_in, const int* in_sizes, int n_in,
                              void* d_out, int out_size, void* d_ws, size_t ws_size,
                              hipStream_t stream) {
  (void)in_sizes; (void)n_in; (void)out_size;
  if (ws_size < WS_NEEDED) return;  // need ~313 MB scratch

  const float* xyz = (const float*)d_in[0];   // [B,3,N]
  const float* pts = (const float*)d_in[1];   // [B,64,N]
  const float* W0  = (const float*)d_in[2];
  const float* g0  = (const float*)d_in[4];
  const float* be0 = (const float*)d_in[5];
  const float* W1  = (const float*)d_in[6];
  const float* g1  = (const float*)d_in[8];
  const float* be1 = (const float*)d_in[9];
  const float* W2  = (const float*)d_in[10];
  const float* g2  = (const float*)d_in[12];
  const float* be2 = (const float*)d_in[13];
  // biases b0/b1/b2 cancel exactly under BN mean subtraction -> unused.

  char* ws = (char*)d_ws;
  float*    newxyz = (float*)(ws + OFF_NEWXYZ);
  _Float16* ptsT   = (_Float16*)(ws + OFF_PTST);
  _Float16* W0t    = (_Float16*)(ws + OFF_W0T);
  _Float16* W1t    = (_Float16*)(ws + OFF_W1T);
  _Float16* W2t    = (_Float16*)(ws + OFF_W2T);
  float*    scb    = (float*)(ws + OFF_SC);
  float *sc0 = scb, *sh0 = scb + 128, *sc1 = scb + 256, *sh1 = scb + 384,
        *sc2 = scb + 512, *sh2 = scb + 640;
  float*    P0 = (float*)(ws + OFF_P0);
  float*    P1 = (float*)(ws + OFF_P1);
  float*    P2 = (float*)(ws + OFF_P2);
  _Float16* A0 = (_Float16*)(ws + OFF_A0);
  _Float16* Y0 = (_Float16*)(ws + OFF_Y0);
  _Float16* Y1 = A0;                       // alias: A0 dead after GEMM0
  _Float16* Y2 = (_Float16*)(ws + OFF_Y2);

  float* out        = (float*)d_out;       // [B,3,S] then [B,128,S]
  float* out_points = out + (size_t)BQ * 3 * SQ;

  fps_kernel<<<BQ, 1024, 0, stream>>>(xyz, newxyz, out);
  transpose_points_kernel<<<(BQ * 64 * NQ) / 256, 256, 0, stream>>>(pts, ptsT);
  prep_weights_kernel<<<8, 256, 0, stream>>>(W0, W1, W2, W0t, W1t, W2t);
  ballquery_gather_kernel<<<(BQ * SQ) / 8, 256, 0, stream>>>(xyz, ptsT, newxyz, A0);

  gemm_bn_kernel<C0P, 64, false><<<GEMM_BLOCKS, 256, 0, stream>>>(A0, W0t, Y0, sc0, sh0, P0, MT);
  finalize_stats_kernel<<<1, 128, 0, stream>>>(P0, GEMM_BLOCKS, 64, g0, be0, sc0, sh0);

  gemm_bn_kernel<64, 64, true><<<GEMM_BLOCKS, 256, 0, stream>>>(Y0, W1t, Y1, sc0, sh0, P1, MT);
  finalize_stats_kernel<<<1, 128, 0, stream>>>(P1, GEMM_BLOCKS, 64, g1, be1, sc1, sh1);

  gemm_bn_kernel<64, 128, true><<<GEMM_BLOCKS, 256, 0, stream>>>(Y1, W2t, Y2, sc1, sh1, P2, MT);
  finalize_stats_kernel<<<1, 128, 0, stream>>>(P2, GEMM_BLOCKS, 128, g2, be2, sc2, sh2);

  maxpool_kernel<<<BQ * 32, 256, 0, stream>>>(Y2, sc2, sh2, out_points);
}
